// AliBiMultiHeadAttention_81114752352876
// MI455X (gfx1250) — matrix operand order repaired and checked by reading, not compiled, not run on hardware
//
#include <hip/hip_runtime.h>

typedef __attribute__((ext_vector_type(16))) _Float16 v16h;
typedef __attribute__((ext_vector_type(8)))  float    v8f;

#define B_  2
#define T_  2048
#define C_  512
#define H_  16
#define D_  32
#define SCALE_ 0.17677669529663687f  /* 1/sqrt(32) */

__device__ __forceinline__ v8f wmma_f16(v16h a, v16h b, v8f c) {
    return __builtin_amdgcn_wmma_f32_16x16x32_f16(
        /*neg_a=*/false, a, /*neg_b=*/false, b,
        /*c_mod=*/(short)0, c, /*reuse_a=*/false, /*reuse_b=*/false);
}

// A-fragment (16x32, 16-bit): lane<16 holds row=lane, K0..15;
// lane>=16 holds row=lane-16, K16..31.
__device__ __forceinline__ v16h load_a_frag(const _Float16* rowp, int laneHi) {
    v16h a;
    const _Float16* p = rowp + laneHi * 16;
#pragma unroll
    for (int i = 0; i < 16; ++i) a[i] = p[i];
    return a;
}

// B-fragment (32x16, 16-bit) where B[k][n] is contiguous along k for fixed n:
// lane = column n (mod 16), lane half selects K 0-15 vs 16-31.
__device__ __forceinline__ v16h load_b_frag_contig(const _Float16* p) {
    v16h b;
#pragma unroll
    for (int i = 0; i < 16; ++i) b[i] = p[i];
    return b;
}

// ---------------------------------------------------------------- convert
__global__ void f32_to_f16_kernel(const float* __restrict__ in,
                                  _Float16* __restrict__ out, int n) {
    int i = blockIdx.x * blockDim.x + threadIdx.x;
    if (i < n) out[i] = (_Float16)in[i];
}

// ---------------------------------------------------------------- GEMM: y = A * W^T + b
// A: (M x K) f16 row-major.  W: (N x K) f16 row-major (so B[k][n] = W[n][k]).
// mode 0: f16 -> (B,H,T,D) (Q,K).  mode 1: f32 row-major (O proj).
// mode 2: f16 -> (B,H,D,T) transposed (V, so PV B-fragments are contiguous).
__global__ void gemm_wmma_kernel(const _Float16* __restrict__ A,
                                 const _Float16* __restrict__ W,
                                 const float* __restrict__ bias,
                                 _Float16* __restrict__ outH,
                                 float* __restrict__ outF,
                                 int M, int N, int K, int mode) {
    const int lane   = threadIdx.x & 31;
    const int wave   = threadIdx.x >> 5;
    const int laneLo = lane & 15;
    const int laneHi = lane >> 4;

    const int tilesN = N >> 4;
    const int tile   = blockIdx.x * 4 + wave;   // 4 waves/block
    const int tm = tile / tilesN, tn = tile % tilesN;
    const int mBase = tm << 4, nBase = tn << 4;

    const _Float16* ap = A + (size_t)(mBase + laneLo) * K;
    const _Float16* wp = W + (size_t)(nBase + laneLo) * K + laneHi * 16;

    // software-pipelined k-loop: fragments for step k+1 load while WMMA k runs
    v8f  acc = {};
    v16h af = load_a_frag(ap, laneHi);
    v16h bf = load_b_frag_contig(wp);
    for (int k0 = 32; k0 < K; k0 += 32) {
        __builtin_prefetch(ap + k0 + 32, 0, 0);     // global_prefetch_b8
        v16h af2 = load_a_frag(ap + k0, laneHi);
        v16h bf2 = load_b_frag_contig(wp + k0);
        acc = wmma_f16(af, bf, acc);
        af = af2; bf = bf2;
    }
    acc = wmma_f16(af, bf, acc);

#pragma unroll
    for (int v = 0; v < 8; ++v) {
        const int row = mBase + v + 8 * laneHi;       // global m
        const int col = nBase + laneLo;               // global n
        const float val = acc[v] + bias[col];
        const int bb = row >> 11;                     // row / T_
        const int t  = row & (T_ - 1);
        const int hh = col >> 5;                      // col / D_
        const int dd = col & (D_ - 1);
        if (mode == 0) {
            outH[(((size_t)bb * H_ + hh) * T_ + t) * D_ + dd] = (_Float16)val;
        } else if (mode == 2) {
            outH[(((size_t)bb * H_ + hh) * D_ + dd) * T_ + t] = (_Float16)val;
        } else {
            outF[(size_t)row * N + col] = val;
        }
    }
}

// ---------------------------------------------------------------- attention
// One block per (b, 16-query-row tile); 8 waves, each owns a contiguous
// 256-key slice. Per head, each wave async-copies its 16KB K-slice into LDS
// once (ASYNCcnt path) and all three passes read fragments from LDS.
__global__ void attn_alibi_kernel(const _Float16* __restrict__ Qh,
                                  const _Float16* __restrict__ Kh,
                                  const _Float16* __restrict__ Vt,
                                  _Float16* __restrict__ attnh,
                                  float* __restrict__ avg) {
    extern __shared__ _Float16 kbuf[];             // [8 waves][256 keys][32 d] = 128 KB
    __shared__ float    red[8][16];
    __shared__ float    m_s[16];
    __shared__ float    l_s[16];
    __shared__ _Float16 pstage[8][16][32];
    __shared__ float    accst[16][32];

    const int tid    = threadIdx.x;
    const int lane   = tid & 31;
    const int wave   = tid >> 5;
    const int laneLo = lane & 15;
    const int laneHi = lane >> 4;

    const int b      = blockIdx.x / (T_ / 16);
    const int qBase  = (blockIdx.x % (T_ / 16)) * 16;
    const int kSlice = wave * 256;                 // this wave's keys [kSlice, kSlice+256)

    _Float16* kw = kbuf + (size_t)wave * 256 * D_;
    const unsigned ldsKBase = (unsigned)(unsigned long long)kw;

    for (int h = 0; h < H_; ++h) {
        const float  slope     = exp2f(-(float)(h + 1) / (float)H_);
        const size_t headBase  = ((size_t)b * H_ + h) * T_;
        const size_t headBaseV = ((size_t)b * H_ + h) * D_ * T_;

        // ---- stage this wave's K slice (contiguous 16 KB) into LDS -----
        {
            const unsigned long long gbase =
                (unsigned long long)(Kh + (headBase + (size_t)kSlice) * D_);
            asm volatile("s_wait_dscnt 0x0" ::: "memory");  // prior-head LDS reads done
#pragma unroll
            for (int j = 0; j < 32; ++j) {                  // 512 B per lane, b128 each
                const unsigned off = (unsigned)((j * 32 + lane) * 16);
                const unsigned           ldst = ldsKBase + off;
                const unsigned long long ga   = gbase + off;
                asm volatile("global_load_async_to_lds_b128 %0, %1, off"
                             :: "v"(ldst), "v"(ga) : "memory");
            }
            asm volatile("s_wait_asynccnt 0x0" ::: "memory");
        }

        // Q fragment (same 16 rows for all waves)
        const _Float16* qp = Qh + (headBase + qBase + laneLo) * D_;
        const v16h qa = load_a_frag(qp, laneHi);

        // ---- pass 1: row max -------------------------------------------
        float rmax[8];
#pragma unroll
        for (int v = 0; v < 8; ++v) rmax[v] = -3.0e38f;
        for (int tt = 0; tt < 16; ++tt) {
            const int keyLoc = tt * 16;
            v8f sc = wmma_f16(qa,
                load_b_frag_contig(kw + (keyLoc + laneLo) * D_ + laneHi * 16), v8f{});
#pragma unroll
            for (int v = 0; v < 8; ++v) {
                const int row = qBase + v + 8 * laneHi;
                const int col = kSlice + keyLoc + laneLo;
                const float s = sc[v] * SCALE_ + slope * (float)(row - col);
                rmax[v] = fmaxf(rmax[v], s);
            }
        }
#pragma unroll
        for (int v = 0; v < 8; ++v)
            for (int off = 8; off >= 1; off >>= 1)
                rmax[v] = fmaxf(rmax[v], __shfl_xor(rmax[v], off, 32));
        if (laneLo == 0) {
#pragma unroll
            for (int v = 0; v < 8; ++v) red[wave][laneHi * 8 + v] = rmax[v];
        }
        __syncthreads();
        if (tid < 16) {
            float mm = -3.0e38f;
            for (int w = 0; w < 8; ++w) mm = fmaxf(mm, red[w][tid]);
            m_s[tid] = mm;
        }
        __syncthreads();

        // ---- pass 2: exp-sum + P*V (unnormalized) ----------------------
        float rsum[8];
#pragma unroll
        for (int v = 0; v < 8; ++v) rsum[v] = 0.f;
        v8f acc0 = {}, acc1 = {};
        for (int c = 0; c < 8; ++c) {
            const int keyLoc = c * 32;                       // within slice
            const int keyG   = kSlice + keyLoc;              // global key
#pragma unroll
            for (int half = 0; half < 2; ++half) {
                const int kl2 = keyLoc + half * 16;
                v8f sc = wmma_f16(qa,
                    load_b_frag_contig(kw + (kl2 + laneLo) * D_ + laneHi * 16), v8f{});
#pragma unroll
                for (int v = 0; v < 8; ++v) {
                    const int rowL = v + 8 * laneHi;
                    const int row  = qBase + rowL;
                    const int col  = kSlice + kl2 + laneLo;
                    const float s = sc[v] * SCALE_ + slope * (float)(row - col);
                    const float p = __expf(s - m_s[rowL]);
                    rsum[v] += p;
                    pstage[wave][rowL][half * 16 + laneLo] = (_Float16)p;
                }
            }
            // per-wave LDS staging turnaround (DS pipeline in-order per wave)
            asm volatile("s_wait_dscnt 0x0" ::: "memory");
            const v16h pa = load_a_frag(&pstage[wave][laneLo][0], laneHi);
            // V transposed (B,H,D,T): contiguous 32B per lane
            const _Float16* vp0 = Vt + headBaseV + (size_t)laneLo * T_ + keyG + laneHi * 16;
            const v16h vb0 = load_b_frag_contig(vp0);
            const v16h vb1 = load_b_frag_contig(vp0 + 16 * (size_t)T_);
            acc0 = wmma_f16(pa, vb0, acc0);
            acc1 = wmma_f16(pa, vb1, acc1);
        }
#pragma unroll
        for (int v = 0; v < 8; ++v)
            for (int off = 8; off >= 1; off >>= 1)
                rsum[v] += __shfl_xor(rsum[v], off, 32);
        if (laneLo == 0) {
#pragma unroll
            for (int v = 0; v < 8; ++v) red[wave][laneHi * 8 + v] = rsum[v];
        }
        __syncthreads();
        if (tid < 16) {
            float ss = 0.f;
            for (int w = 0; w < 8; ++w) ss += red[w][tid];
            l_s[tid] = ss;
        }
        for (int i = tid; i < 512; i += 256) ((float*)accst)[i] = 0.f;
        __syncthreads();
#pragma unroll
        for (int v = 0; v < 8; ++v) {
            const int rowL = v + 8 * laneHi;
            atomicAdd(&accst[rowL][laneLo],      acc0[v]);   // ds_add_f32
            atomicAdd(&accst[rowL][laneLo + 16], acc1[v]);
        }
        __syncthreads();
        for (int i = tid; i < 512; i += 256) {
            const int rowL = i >> 5, d = i & 31;
            const float val = accst[rowL][d] / l_s[rowL];
            attnh[((size_t)b * T_ + qBase + rowL) * C_ + h * D_ + d] = (_Float16)val;
        }

        // ---- pass 3: normalized probs -> avg over heads ---------------
        for (int tt = 0; tt < 16; ++tt) {
            const int keyLoc = tt * 16;
            v8f sc = wmma_f16(qa,
                load_b_frag_contig(kw + (keyLoc + laneLo) * D_ + laneHi * 16), v8f{});
#pragma unroll
            for (int v = 0; v < 8; ++v) {
                const int rowL = v + 8 * laneHi;
                const int row  = qBase + rowL;
                const int col  = kSlice + keyLoc + laneLo;
                const float s = sc[v] * SCALE_ + slope * (float)(row - col);
                const float p = (__expf(s - m_s[rowL]) / l_s[rowL]) * (1.0f / (float)H_);
                float* ap = avg + ((size_t)b * T_ + row) * T_ + col;
                if (h == 0) *ap = p; else *ap += p;
            }
        }
        __syncthreads();   // protect m_s/l_s/red reuse next head
    }
}

// ----------------------------------------------------------------
extern "C" void kernel_launch(void* const* d_in, const int* in_sizes, int n_in,
                              void* d_out, int out_size, void* d_ws, size_t ws_size,
                              hipStream_t stream) {
    const float* x  = (const float*)d_in[0];
    const float* Wq = (const float*)d_in[1];
    const float* bq = (const float*)d_in[2];
    const float* Wk = (const float*)d_in[3];
    const float* bk = (const float*)d_in[4];
    const float* Wv = (const float*)d_in[5];
    const float* bv = (const float*)d_in[6];
    const float* Wo = (const float*)d_in[7];
    const float* bo = (const float*)d_in[8];

    float* out = (float*)d_out;                       // (B,T,C)
    float* avg = out + (size_t)B_ * T_ * C_;          // (B,T,T)

    const int M = B_ * T_;                            // 4096
    char* ws = (char*)d_ws;
    _Float16* xh    = (_Float16*)ws;  ws += (size_t)M * C_ * 2;
    _Float16* wqh   = (_Float16*)ws;  ws += (size_t)C_ * C_ * 2;
    _Float16* wkh   = (_Float16*)ws;  ws += (size_t)C_ * C_ * 2;
    _Float16* wvh   = (_Float16*)ws;  ws += (size_t)C_ * C_ * 2;
    _Float16* woh   = (_Float16*)ws;  ws += (size_t)C_ * C_ * 2;
    _Float16* Qh    = (_Float16*)ws;  ws += (size_t)B_ * H_ * T_ * D_ * 2;
    _Float16* Kh    = (_Float16*)ws;  ws += (size_t)B_ * H_ * T_ * D_ * 2;
    _Float16* Vt    = (_Float16*)ws;  ws += (size_t)B_ * H_ * T_ * D_ * 2;  // (B,H,D,T)
    _Float16* attnh = (_Float16*)ws;  ws += (size_t)M * C_ * 2;

    const int nx = M * C_, nw = C_ * C_;
    f32_to_f16_kernel<<<(nx + 255) / 256, 256, 0, stream>>>(x,  xh,  nx);
    f32_to_f16_kernel<<<(nw + 255) / 256, 256, 0, stream>>>(Wq, wqh, nw);
    f32_to_f16_kernel<<<(nw + 255) / 256, 256, 0, stream>>>(Wk, wkh, nw);
    f32_to_f16_kernel<<<(nw + 255) / 256, 256, 0, stream>>>(Wv, wvh, nw);
    f32_to_f16_kernel<<<(nw + 255) / 256, 256, 0, stream>>>(Wo, woh, nw);

    const int tiles = (M / 16) * (C_ / 16);           // 8192 tiles, 4 waves/block
    gemm_wmma_kernel<<<tiles / 4, 128, 0, stream>>>(xh, wqh, bq, Qh, nullptr, M, C_, C_, 0);
    gemm_wmma_kernel<<<tiles / 4, 128, 0, stream>>>(xh, wkh, bk, Kh, nullptr, M, C_, C_, 0);
    gemm_wmma_kernel<<<tiles / 4, 128, 0, stream>>>(xh, wvh, bv, Vt, nullptr, M, C_, C_, 2);

    const size_t attnLds = (size_t)8 * 256 * D_ * sizeof(_Float16);   // 128 KB dynamic LDS
    attn_alibi_kernel<<<B_ * (T_ / 16), 256, attnLds, stream>>>(Qh, Kh, Vt, attnh, avg);

    gemm_wmma_kernel<<<tiles / 4, 128, 0, stream>>>(attnh, woh, bo, nullptr, out, M, C_, C_, 1);
}
